// RecursiveSimplePtreeLayer_53807350284927
// MI455X (gfx1250) — matrix operand order repaired and checked by reading, not compiled, not run on hardware
//
#include <hip/hip_runtime.h>
#include <hip/hip_bf16.h>

typedef __attribute__((ext_vector_type(16))) __bf16 v16bf;
typedef __attribute__((ext_vector_type(8)))  float  v8f;

#define HD 256      // hidden dim
#define NT 8192     // trees
#define CC 16       // children
#define AP 264      // A LDS pitch (ushorts): 256 + pad (16B aligned, bank-spread)
#define BPQ 264     // stage1 B LDS pitch (ushorts): 256 K-vals + pad
#define BP2 264     // stage2 B LDS pitch

__device__ __forceinline__ unsigned short f2bf(float f) {
  unsigned int u = __float_as_uint(f);
  unsigned int r = u + 0x7FFFu + ((u >> 16) & 1u);   // round-to-nearest-even
  return (unsigned short)(r >> 16);
}

union AF { v16bf v; uint4 q[2]; };

// ---------------------------------------------------------------------------
// prep: bf16-convert + transpose weights; bsum = sum_k bz[k]
//   Wt  [256][1024] : Wt[d][k*256+h]  = Wz[k][h][d]
//   Wzft[256][256]  : Wzft[d][h]      = Wzf[h][d]
// ---------------------------------------------------------------------------
__global__ __launch_bounds__(256)
void prep_kernel(const float* __restrict__ Wz, const float* __restrict__ bz,
                 const float* __restrict__ Wzf,
                 unsigned short* __restrict__ Wt, unsigned short* __restrict__ Wzft,
                 float* __restrict__ bsum) {
  int i = blockIdx.x * 256 + threadIdx.x;          // 0 .. 262143
  int d  = i >> 10;
  int kh = i & 1023;
  Wt[i] = f2bf(Wz[kh * 256 + d]);
  if (i < 65536) {
    int dd = i >> 8, hh = i & 255;
    Wzft[i] = f2bf(Wzf[hh * 256 + dd]);
  }
  if (i < 256) bsum[i] = bz[i] + bz[256 + i] + bz[512 + i] + bz[768 + i];
}

// ---------------------------------------------------------------------------
// stage1: per tree t: rolled = sum_k roll(D_t,k) @ Wz[k]  ([16x1024]@[1024x256])
//         S[t,:] = sum_c elu(rolled[c,:] + bsum)          (bf16)
// 4 waves / block, one tree per wave, 2 N-tiles per wave iteration
// (A fragment shared by both WMMAs -> 3 ds_load_b128 per wmma).
// B staged per (N-pair, K-quarter); K-quarter == one Wz layer, so the
// cyclic-roll row index is loop-invariant inside the K loop.
// ---------------------------------------------------------------------------
__global__ __launch_bounds__(128)
void stage1_kernel(const float* __restrict__ x, const int* __restrict__ leaf_idx,
                   const unsigned short* __restrict__ Wt, const float* __restrict__ bsum,
                   unsigned short* __restrict__ S) {
  __shared__ unsigned short Alds[4 * 16 * AP];   // 33,792 B
  __shared__ unsigned short Blds[32 * BPQ];      // 16,896 B

  const int tid  = threadIdx.x;
  const int lane = tid & 31;
  const int w    = tid >> 5;
  const int t    = blockIdx.x * 4 + w;

  // ---- gather A: 16 rows x 256 f32 -> bf16 LDS (row-major, pitch AP) ----
  const long tb = (long)t * CC;
  for (int c = 0; c < 16; ++c) {
    const int row = leaf_idx[tb + c];
    const float4* rp = reinterpret_cast<const float4*>(x + (long)row * HD);
    unsigned short* dst = &Alds[w * 16 * AP + c * AP];
#pragma unroll
    for (int i = 0; i < 2; ++i) {
      float4 v = rp[lane + i * 32];
      unsigned int p0 = (unsigned int)f2bf(v.x) | ((unsigned int)f2bf(v.y) << 16);
      unsigned int p1 = (unsigned int)f2bf(v.z) | ((unsigned int)f2bf(v.w) << 16);
      uint2 pk; pk.x = p0; pk.y = p1;
      *reinterpret_cast<uint2*>(dst + lane * 4 + i * 128) = pk;
    }
  }

  const int m  = lane & 15;     // tile row (M=c) / tile column (N) index
  const int hi = lane >> 4;
  const int ab = hi * 8;        // A-fragment K base
  const int bb = hi * 8;        // B-fragment K base
  const int sj = tid >> 2;      // staging: B row (N column) 0..31
  const int sp = tid & 3;       // staging: 128B part within row

  for (int ng = 0; ng < 8; ++ng) {
    const int n0 = ng * 32;
    v8f acc0 = {0.f, 0.f, 0.f, 0.f, 0.f, 0.f, 0.f, 0.f};
    v8f acc1 = {0.f, 0.f, 0.f, 0.f, 0.f, 0.f, 0.f, 0.f};

    for (int qq = 0; qq < 4; ++qq) {           // K-quarter == Wz layer qq
      __syncthreads();
      { // stage B chunk: Blds[j][kr] = Wt[n0+j][qq*256 + kr], kr<256
        const uint4* src = reinterpret_cast<const uint4*>(
            Wt + (long)(n0 + sj) * 1024 + qq * 256 + sp * 64);
        uint4* dstq = reinterpret_cast<uint4*>(&Blds[sj * BPQ + sp * 64]);
#pragma unroll
        for (int q = 0; q < 8; ++q) dstq[q] = src[q];
      }
      __syncthreads();

      { // prefetch next B chunk into L2/WGP$ while we compute on this one
        int nqq = qq + 1, nng = ng;
        if (nqq == 4) { nqq = 0; nng = ng + 1; }
        if (nng < 8) {
          __builtin_prefetch(Wt + (long)(nng * 32 + sj) * 1024 + nqq * 256 + sp * 64, 0, 1);
        }
      }

      const int r = (m - qq) & 15;                         // cyclic roll (hoisted)
      const unsigned short* abase = &Alds[w * 16 * AP + r * AP];
#pragma unroll
      for (int kk2 = 0; kk2 < 8; ++kk2) {
        const int aoff = (kk2 << 5) + ab;
        AF a, b0, b1;
        a.q[0] = *reinterpret_cast<const uint4*>(abase + aoff);
        a.q[1] = *reinterpret_cast<const uint4*>(abase + aoff + 16);
        const unsigned short* bp0 = &Blds[m * BPQ + (kk2 << 5) + bb];
        const unsigned short* bp1 = bp0 + 16 * BPQ;
        b0.q[0] = *reinterpret_cast<const uint4*>(bp0);
        b0.q[1] = *reinterpret_cast<const uint4*>(bp0 + 16);
        b1.q[0] = *reinterpret_cast<const uint4*>(bp1);
        b1.q[1] = *reinterpret_cast<const uint4*>(bp1 + 16);
        acc0 = __builtin_amdgcn_wmma_f32_16x16x32_bf16(false, a.v, false, b0.v,
                                                       (short)0, acc0, false, false);
        acc1 = __builtin_amdgcn_wmma_f32_16x16x32_bf16(false, a.v, false, b1.v,
                                                       (short)0, acc1, false, false);
      }
    }

    // epilogue: +bias, ELU, sum over all 16 children, write both S tile rows
#pragma unroll
    for (int q = 0; q < 2; ++q) {
      v8f accq = q ? acc1 : acc0;
      const float bs = bsum[n0 + q * 16 + m];
      float partial = 0.f;
#pragma unroll
      for (int j = 0; j < 8; ++j) {
        float v = accq[j] + bs;
        partial += (v > 0.f) ? v : (__expf(v) - 1.f);
      }
      partial += __shfl_xor(partial, 16, 32);    // fold M=0..7 with M=8..15
      if (lane < 16) S[(long)t * HD + n0 + q * 16 + lane] = f2bf(partial);
    }
  }
}

// ---------------------------------------------------------------------------
// stage2: out = elu( S[8192x256] @ Wzf + 16*bzf )   (f32 out)
// 4 waves / block, one 16-row M-tile per wave.
// ---------------------------------------------------------------------------
__global__ __launch_bounds__(128)
void stage2_kernel(const unsigned short* __restrict__ S,
                   const unsigned short* __restrict__ Wzft,
                   const float* __restrict__ bzf, float* __restrict__ out) {
  __shared__ unsigned short Alds[4 * 16 * AP];   // 33,792 B
  __shared__ unsigned short Blds[16 * BP2];      //  8,448 B

  const int tid  = threadIdx.x;
  const int lane = tid & 31;
  const int w    = tid >> 5;
  const int mt   = blockIdx.x * 4 + w;           // 16-row tile of S

  // stage A (already bf16): 16 rows x 256
  for (int c = 0; c < 16; ++c) {
    const uint4* src = reinterpret_cast<const uint4*>(S + (long)(mt * 16 + c) * HD) + lane;
    *reinterpret_cast<uint4*>(&Alds[w * 16 * AP + c * AP + lane * 8]) = *src;
  }

  const int m  = lane & 15;
  const int hi = lane >> 4;
  const int ab = hi * 8;
  const int bb = hi * 8;

  for (int nt = 0; nt < 16; ++nt) {
    const int n0 = nt * 16;
    __syncthreads();
    { // stage B: Blds[j][h] = Wzft[n0+j][h], h<256
      int j = tid >> 3, part = tid & 7;
      const uint4* src = reinterpret_cast<const uint4*>(Wzft + (long)(n0 + j) * 256 + part * 32);
      uint4* dstq = reinterpret_cast<uint4*>(&Blds[j * BP2 + part * 32]);
#pragma unroll
      for (int q = 0; q < 4; ++q) dstq[q] = src[q];
    }
    __syncthreads();

    v8f acc = {0.f, 0.f, 0.f, 0.f, 0.f, 0.f, 0.f, 0.f};
#pragma unroll
    for (int kk = 0; kk < 8; ++kk) {
      AF a, b;
      const unsigned short* ap = &Alds[w * 16 * AP + m * AP + (kk << 5) + ab];
      a.q[0] = *reinterpret_cast<const uint4*>(ap);
      a.q[1] = *reinterpret_cast<const uint4*>(ap + 16);
      const unsigned short* bp = &Blds[m * BP2 + (kk << 5) + bb];
      b.q[0] = *reinterpret_cast<const uint4*>(bp);
      b.q[1] = *reinterpret_cast<const uint4*>(bp + 16);
      acc = __builtin_amdgcn_wmma_f32_16x16x32_bf16(false, a.v, false, b.v,
                                                    (short)0, acc, false, false);
    }

    const float bias = 16.0f * bzf[n0 + m];
#pragma unroll
    for (int j = 0; j < 8; ++j) {
      float v = acc[j] + bias;
      float e = (v > 0.f) ? v : (__expf(v) - 1.f);
      out[(long)(mt * 16 + j + hi * 8) * HD + n0 + m] = e;
    }
  }
}

// ---------------------------------------------------------------------------
extern "C" void kernel_launch(void* const* d_in, const int* in_sizes, int n_in,
                              void* d_out, int out_size, void* d_ws, size_t ws_size,
                              hipStream_t stream) {
  const float* x    = (const float*)d_in[0];
  const float* Wz   = (const float*)d_in[1];
  const float* bz   = (const float*)d_in[2];
  const float* Wzf  = (const float*)d_in[3];
  const float* bzf  = (const float*)d_in[4];
  const int*   leaf = (const int*)d_in[5];

  // workspace layout
  unsigned short* Wt   = (unsigned short*)d_ws;        // 256*1024 bf16  (512 KB)
  unsigned short* Wzft = Wt + 256 * 1024;              // 256*256  bf16  (128 KB)
  float*          bsum = (float*)(Wzft + 256 * 256);   // 256 f32
  unsigned short* S    = (unsigned short*)(bsum + 256);// 8192*256 bf16  (4 MB)

  float* out = (float*)d_out;

  prep_kernel  <<<1024, 256, 0, stream>>>(Wz, bz, Wzf, Wt, Wzft, bsum);
  stage1_kernel<<<NT / 4, 128, 0, stream>>>(x, leaf, Wt, bsum, S);
  stage2_kernel<<<(NT / 16) / 4, 128, 0, stream>>>(S, Wzft, bzf, out);
}
